// ToRGBModule_61881888801471
// MI455X (gfx1250) — matrix operand order repaired and checked by reading, not compiled, not run on hardware
//
#include <hip/hip_runtime.h>
#include <hip/hip_bf16.h>
#include <math.h>

typedef __attribute__((ext_vector_type(16))) __bf16 v16bf;
typedef __attribute__((ext_vector_type(8)))  float  v8f;

#define BATCH 2
#define C_CH  128
#define WDIM  512
#define SIN   256
#define SOUT  258            // conv output size (pad 2, k=3)
#define SPAD  260            // padded conv input
#define NPOS  (SOUT*SOUT)    // 66564 positions per batch
#define KTOT  (9*C_CH)       // 1152
#define NTILE 64
#define NUM_NTILES ((NPOS + NTILE - 1)/NTILE)  // 1041
#define NCHUNKS (KTOT/32)    // 36
#define LSTR  40             // padded LDS row stride (bf16 elems) for bank spread

union Frag { v16bf v; uint4 q[2]; };

// ---------------------------------------------------------------- k0: styles
__global__ void k0_styles(const float* __restrict__ w, const float* __restrict__ afw,
                          const float* __restrict__ afb, const float* __restrict__ arw,
                          const float* __restrict__ arb, const float* __restrict__ wrgb,
                          const float* __restrict__ rgb_ema,
                          float* __restrict__ s_feat_norm, float* __restrict__ coeff)
{
  __shared__ float red[256];
  int tid = threadIdx.x;
  int b = tid >> 7, c = tid & 127;
  const float wgain = 0.04419417382415922f;   // 1/sqrt(512)
  const float* wrow = w + b*WDIM;
  const float* arow = afw + c*WDIM;
  float acc = 0.f;
  for (int j = 0; j < WDIM; ++j) acc += wrow[j]*arow[j];
  float s = acc * wgain + afb[c];
  red[tid] = s*s; __syncthreads();
  for (int off = 128; off > 0; off >>= 1) { if (tid < off) red[tid] += red[tid+off]; __syncthreads(); }
  float snorm = s * rsqrtf(red[0] * (1.0f/256.0f));
  s_feat_norm[tid] = snorm;
  // RGB style: no mean-norm; scale by 1/sqrt(C) and rsqrt(rgb_ema)
  const float* arow2 = arw + c*WDIM;
  float acc2 = 0.f;
  for (int j = 0; j < WDIM; ++j) acc2 += wrow[j]*arow2[j];
  float sr = (acc2*wgain + arb[c]) * 0.08838834764831845f * rsqrtf(rgb_ema[0]);
  for (int o = 0; o < 3; ++o)
    coeff[(b*3+o)*C_CH + c] = wrgb[o*C_CH + c] * sr;
}

// ------------------------------------------- k1: modulated weights -> bf16 A
// A layout: [b][o][k] with k = tap*128 + ci  (tap = kh*3+kw)
__global__ void k1_modweights(const float* __restrict__ weight,
                              const float* __restrict__ s_feat_norm,
                              const float* __restrict__ feat_ema,
                              __bf16* __restrict__ A)
{
  __shared__ float red[256];
  int tid = threadIdx.x;
  int bo = blockIdx.x;                // b*128 + o
  int b = bo >> 7, o = bo & 127;
  const float* wo = weight + (size_t)o * KTOT;   // [o][ci][kh][kw]: o*1152 + ci*9 + tap
  float ss = 0.f;
  for (int k = tid; k < KTOT; k += 256) { float v = wo[k]; ss += v*v; }
  red[tid] = ss; __syncthreads();
  for (int off = 128; off > 0; off >>= 1) { if (tid < off) red[tid] += red[tid+off]; __syncthreads(); }
  float wrms_inv = rsqrtf(red[0] * (1.0f/KTOT));
  __syncthreads();
  const float* sn = s_feat_norm + b*C_CH;
  float ss2 = 0.f;
  for (int k = tid; k < KTOT; k += 256) {
    int ci = k & 127, tap = k >> 7;
    float v = wo[ci*9 + tap] * wrms_inv * sn[ci];
    ss2 += v*v;
  }
  red[tid] = ss2; __syncthreads();
  for (int off = 128; off > 0; off >>= 1) { if (tid < off) red[tid] += red[tid+off]; __syncthreads(); }
  float d = rsqrtf(red[0] + 1e-8f) * rsqrtf(feat_ema[0]);
  __bf16* Arow = A + (size_t)bo * KTOT;
  for (int k = tid; k < KTOT; k += 256) {
    int ci = k & 127, tap = k >> 7;
    Arow[k] = (__bf16)(wo[ci*9 + tap] * wrms_inv * sn[ci] * d);
  }
}

// ------------------------------------------------ k2: pad + convert to bf16
__global__ void k2_pad(const float* __restrict__ x, __bf16* __restrict__ xpad, int total)
{
  int gid = blockIdx.x*blockDim.x + threadIdx.x;
  if (gid >= total) return;
  int wp = gid % SPAD; int t = gid / SPAD;
  int hp = t % SPAD;   int bc = t / SPAD;
  int h = hp - 2, ww = wp - 2;
  float v = 0.f;
  if (h >= 0 && h < SIN && ww >= 0 && ww < SIN)
    v = x[((size_t)bc*SIN + h)*SIN + ww];
  xpad[gid] = (__bf16)v;
}

// --------------------------------- k3: implicit-GEMM conv via WMMA bf16
// Workgroup: 256 thr = 8 waves. WG tile: M=128 (all out ch) x N=64 positions.
// Wave (wv&3, wv>>2) owns 32x32 -> 2x2 WMMA 16x16 accumulators.
__global__ void __launch_bounds__(256) k3_conv_wmma(const __bf16* __restrict__ A,
                                                    const __bf16* __restrict__ xpad,
                                                    const float* __restrict__ bias,
                                                    float* __restrict__ out)
{
  __shared__ __attribute__((aligned(16))) __bf16 As[C_CH*LSTR];   // [m][kk], padded stride
  __shared__ __attribute__((aligned(16))) __bf16 Bs[NTILE*LSTR];  // [n][kk], padded stride
  int tid  = threadIdx.x;
  int lane = tid & 31, wv = tid >> 5;
  int ntile = blockIdx.x, b = blockIdx.y;
  int M0 = (wv & 3) * 32, N0 = (wv >> 2) * 32;
  int lm = lane & 15, lh = lane >> 4;
  v8f acc[2][2] = {};
  const uint4*  Ag = (const uint4*)(A + (size_t)b*C_CH*KTOT);
  const __bf16* xb = xpad + (size_t)b*C_CH*SPAD*SPAD;

  // B-gather bookkeeping: this thread always covers n = n0..n0+7 at channel-lane kk
  int kkB = (tid*8) >> 6;
  int n0  = (tid*8) & 63;
  int ng0 = ntile*NTILE + n0;
  int h0  = ng0 / SOUT, w0 = ng0 % SOUT;   // one div/mod for the whole kernel

  for (int ch = 0; ch < NCHUNKS; ++ch) {
    int kc0 = ch*32;
    int tap = ch >> 2;               // 4 chunks of 32 per tap (128 ch)
    int ci0 = (ch & 3) * 32;
    int kh = tap/3, kw = tap%3;
    // A chunk: 128x32 bf16 = 512 uint4, 2 per thread (16B global -> 16B LDS)
    #pragma unroll
    for (int q2 = 0; q2 < 2; ++q2) {
      int q = tid*2 + q2;
      int m = q >> 2;
      int kk = (q & 3) * 8;
      *(uint4*)&As[m*LSTR + kk] = Ag[m*(KTOT/8) + (kc0 + kk)/8];
    }
    // B chunk: im2col gather; incremental addressing, wrap adds SPAD-SOUT+1
    {
      int h = h0, ww = w0;
      const __bf16* src = xb + ((size_t)(ci0+kkB)*SPAD + (h+kh))*SPAD + (ww+kw);
      #pragma unroll
      for (int e = 0; e < 8; ++e) {
        __bf16 v = (__bf16)0.0f;
        if (ng0 + e < NPOS) v = *src;
        Bs[(n0+e)*LSTR + kkB] = v;
        if (++ww == SOUT) { ww = 0; src += (SPAD - SOUT + 1); }
        else              { ++src; }
      }
    }
    __syncthreads();
    // A frag: lane = M, half-lane selects K groups; two contiguous 16B runs
    Frag af[2], bfr[2];
    #pragma unroll
    for (int ms = 0; ms < 2; ++ms) {
      int m = M0 + ms*16 + lm;
      af[ms].q[0] = *(const uint4*)&As[m*LSTR + lh*8];        // K = lh*8 .. +7
      af[ms].q[1] = *(const uint4*)&As[m*LSTR + 16 + lh*8];   // K = 16+lh*8 .. +7
    }
    // B frag: lane = N, half-lane selects K groups; two contiguous 16B runs
    #pragma unroll
    for (int ns = 0; ns < 2; ++ns) {
      int n = N0 + ns*16 + lm;
      bfr[ns].q[0] = *(const uint4*)&Bs[n*LSTR + lh*8];       // K = lh*8 .. +7
      bfr[ns].q[1] = *(const uint4*)&Bs[n*LSTR + 16 + lh*8];  // K = 16+lh*8 .. +7
    }
    #pragma unroll
    for (int ms = 0; ms < 2; ++ms)
      #pragma unroll
      for (int ns = 0; ns < 2; ++ns)
        acc[ms][ns] = __builtin_amdgcn_wmma_f32_16x16x32_bf16(
            false, af[ms].v, false, bfr[ns].v, (short)0, acc[ms][ns], false, false);
    __syncthreads();
    if (ch + 1 < NCHUNKS)
      __builtin_prefetch(&Ag[(kc0 + 32)/8], 0, 1);   // global_prefetch_b8
  }
  // epilogue: C/D layout N = lane%16, M = vgpr + 8*(lane>=16); fold bias_feat.
  // Position depends only on (ns, lm): one div/mod per ns; out-channel strides NPOS.
  #pragma unroll
  for (int ns = 0; ns < 2; ++ns) {
    int pos = ntile*NTILE + N0 + ns*16 + lm;
    if (pos < NPOS) {
      int h = pos / SOUT, ww = pos % SOUT;
      #pragma unroll
      for (int ms = 0; ms < 2; ++ms) {
        int oc0 = M0 + ms*16 + lh*8;
        float* po = out + ((size_t)(b*C_CH + oc0))*NPOS + h*SOUT + ww;
        #pragma unroll
        for (int r = 0; r < 8; ++r)
          po[(size_t)r*NPOS] = acc[ms][ns][r] + bias[oc0 + r];
      }
    }
  }
}

// -------------------- k4: fused up(2x,12tap) -> lrelu/clamp -> down(2x,12tap)
// 32x32 output tile per WG; all intermediates in LDS (no 522x522 planes).
__global__ void __launch_bounds__(256) k4_updown(const float* __restrict__ conv_out,
                                                 const float* __restrict__ upf,
                                                 const float* __restrict__ dnf,
                                                 float* __restrict__ outp)
{
  __shared__ float zt[42*43];
  __shared__ float u1[74*43];
  __shared__ float u2[74*75];
  __shared__ float t3[32*75];
  int tid = threadIdx.x;
  int plane = blockIdx.y;                       // b*128+ci
  int ty0 = (int)(blockIdx.x >> 3) * 32;
  int tx0 = (int)(blockIdx.x & 7) * 32;
  float fu[12], fd[12];
  #pragma unroll
  for (int t = 0; t < 12; ++t) { fu[t] = upf[11-t]*2.0f; fd[t] = dnf[11-t]; }

  const float* zp = conv_out + (size_t)plane*SOUT*SOUT;
  for (int idx = tid; idx < 42*42; idx += 256) {      // z window (incl. FIR pads)
    int r = idx / 42, c = idx % 42;
    int gh = ty0 - 4 + r, gw = tx0 - 4 + c;
    float v = 0.f;
    if (gh >= 0 && gh < SOUT && gw >= 0 && gw < SOUT) v = zp[gh*SOUT + gw];
    zt[r*43 + c] = v;
  }
  __syncthreads();
  for (int idx = tid; idx < 74*42; idx += 256) {      // upsample along H
    int oh = idx / 42, c = idx % 42;
    int t0 = 1 - (oh & 1);                            // parity: 6 live taps
    float a = 0.f;
    #pragma unroll
    for (int j = 0; j < 6; ++j) {
      int t = t0 + 2*j;
      a += fu[t] * zt[((oh + t - 1) >> 1)*43 + c];
    }
    u1[oh*43 + c] = a;
  }
  __syncthreads();
  for (int idx = tid; idx < 74*74; idx += 256) {      // upsample along W + lrelu
    int oh = idx / 74, ow = idx % 74;
    int t0 = 1 - (ow & 1);
    float a = 0.f;
    #pragma unroll
    for (int j = 0; j < 6; ++j) {
      int t = t0 + 2*j;
      a += fu[t] * u1[oh*43 + ((ow + t - 1) >> 1)];
    }
    a = (a < 0.f ? a*0.2f : a) * 1.4142135623730951f;
    a = fminf(fmaxf(a, -256.f), 256.f);
    u2[oh*75 + ow] = a;
  }
  __syncthreads();
  for (int idx = tid; idx < 32*74; idx += 256) {      // downsample along H
    int mm = idx / 74, ow = idx % 74;
    float a = 0.f;
    #pragma unroll
    for (int t = 0; t < 12; ++t) a += fd[t] * u2[(2*mm + t)*75 + ow];
    t3[mm*75 + ow] = a;
  }
  __syncthreads();
  float* op = outp + (size_t)plane*SIN*SIN;
  for (int idx = tid; idx < 32*32; idx += 256) {      // downsample along W
    int mm = idx / 32, nn = idx % 32;
    float a = 0.f;
    #pragma unroll
    for (int t = 0; t < 12; ++t) a += fd[t] * t3[mm*75 + 2*nn + t];
    op[(ty0 + mm)*SIN + tx0 + nn] = a;
  }
}

// ------------------------------------------------------------- k5: ToRGB
__global__ void k5_torgb(const float* __restrict__ feat, const float* __restrict__ coeff,
                         const float* __restrict__ brgb, float* __restrict__ y, int total)
{
  int gid = blockIdx.x*blockDim.x + threadIdx.x;
  if (gid >= total) return;
  int hw = gid % (SIN*SIN);
  int t  = gid / (SIN*SIN);
  int o = t % 3, b = t / 3;
  const float* cf = coeff + (b*3 + o)*C_CH;
  const float* fp = feat + (size_t)b*C_CH*SIN*SIN + hw;
  float a = 0.f;
  for (int ci = 0; ci < C_CH; ++ci) a += cf[ci] * fp[(size_t)ci*SIN*SIN];
  a += brgb[o];
  y[gid] = fminf(fmaxf(a, -256.f), 256.f);
}

extern "C" void kernel_launch(void* const* d_in, const int* in_sizes, int n_in,
                              void* d_out, int out_size, void* d_ws, size_t ws_size,
                              hipStream_t stream)
{
  (void)in_sizes; (void)n_in; (void)out_size; (void)ws_size;
  const float* x     = (const float*)d_in[0];
  const float* w     = (const float*)d_in[1];
  const float* afw   = (const float*)d_in[2];
  const float* afb   = (const float*)d_in[3];
  const float* wfeat = (const float*)d_in[4];
  const float* bfeat = (const float*)d_in[5];
  const float* arw   = (const float*)d_in[6];
  const float* arb   = (const float*)d_in[7];
  const float* wrgb  = (const float*)d_in[8];
  const float* brgb  = (const float*)d_in[9];
  const float* upf   = (const float*)d_in[10];
  const float* dnf   = (const float*)d_in[11];
  const float* fema  = (const float*)d_in[12];
  const float* rema  = (const float*)d_in[13];

  char* ws = (char*)d_ws;
  float*  s_feat = (float*)(ws + 0);                         //   256 f32
  float*  coeff  = (float*)(ws + 1024);                      //   768 f32
  __bf16* A      = (__bf16*)(ws + 4096);                     //  2*128*1152 bf16
  __bf16* xpad   = (__bf16*)(ws + 4096 + 589824);            //  2*128*260*260 bf16
  float*  convo  = (float*)(ws + 35205120);                  //  2*128*258*258 f32
  float*  featds = (float*)(ws + 103374848);                 //  2*128*256*256 f32

  k0_styles<<<1, 256, 0, stream>>>(w, afw, afb, arw, arb, wrgb, rema, s_feat, coeff);
  k1_modweights<<<BATCH*C_CH, 256, 0, stream>>>(wfeat, s_feat, fema, A);

  int padtot = BATCH*C_CH*SPAD*SPAD;
  k2_pad<<<(padtot + 255)/256, 256, 0, stream>>>(x, xpad, padtot);

  dim3 g3(NUM_NTILES, BATCH);
  k3_conv_wmma<<<g3, 256, 0, stream>>>(A, xpad, bfeat, convo);

  dim3 g4(64, BATCH*C_CH);
  k4_updown<<<g4, 256, 0, stream>>>(convo, upf, dnf, featds);

  int tot5 = BATCH*3*SIN*SIN;
  k5_torgb<<<(tot5 + 255)/256, 256, 0, stream>>>(featds, coeff, brgb, (float*)d_out, tot5);
}
